// BlockSparseAttention_13511967113469
// MI455X (gfx1250) — compile-verified, matrix operand order checked by reading, not run on hardware
//
#include <hip/hip_runtime.h>
#include <hip/hip_bf16.h>

typedef __attribute__((ext_vector_type(16))) __bf16 v16bf;
typedef __attribute__((ext_vector_type(8)))  float  v8f;

#define DIMX   1024
#define HEADS  16
#define BLKSZ  129
#define NB     32
#define DK     64
#define BATCH  4
#define NTOK   (BLKSZ * NB)        // 4128
#define MTOK   (BATCH * NTOK)      // 16512
#define QKV_LD 3072
#define KPAD   160                 // 129 keys padded to 5 * 32

// ---------------------------------------------------------------------------
// f32 -> bf16 elementwise cast
// ---------------------------------------------------------------------------
__global__ __launch_bounds__(256)
void cast_f32_bf16(const float* __restrict__ src, __bf16* __restrict__ dst, size_t n) {
    size_t stride = (size_t)gridDim.x * blockDim.x;
    for (size_t i = (size_t)blockIdx.x * blockDim.x + threadIdx.x; i < n; i += stride)
        dst[i] = (__bf16)src[i];
}

// ---------------------------------------------------------------------------
// C[M,N] = A[M,K] @ B[N,K]^T   (bf16 inputs, f32 accumulate via WMMA)
// Cb != null -> store bf16 ; else Cf (f32) with optional bias.
//
// Workgroup: 256 threads = 8 waves (2 in M x 4 in N) -> 128(M) x 256(N) tile.
// Each wave owns 64x64 = 4x4 WMMA fragments -> 16 v_wmma per K-step against
// 8 fragment loads. LDS double-buffered (48 KB); next K-slice is fetched into
// registers before the current slice's WMMAs, one barrier per K-step.
// ---------------------------------------------------------------------------
__global__ __launch_bounds__(256)
void gemm_bf16_wmma(const __bf16* __restrict__ A, const __bf16* __restrict__ B,
                    float* __restrict__ Cf, __bf16* __restrict__ Cb,
                    const float* __restrict__ bias, int M, int N, int K) {
    __shared__ __align__(32) __bf16 sA[2][128][32];   // 16 KB
    __shared__ __align__(32) __bf16 sB[2][256][32];   // 32 KB

    const int tid  = threadIdx.x;
    const int lane = tid & 31;
    const int wid  = tid >> 5;
    const int wm   = wid & 1;        // 2 waves in M
    const int wn   = wid >> 1;       // 4 waves in N
    const int m0   = blockIdx.x * 128;
    const int n0   = blockIdx.y * 256;

    const int row = lane & 15;
    const int kq  = (lane >> 4) * 16;

    // staging assignment: A 128x32 -> 1 v16bf/thread; B 256x32 -> 2 v16bf/thread
    const int ar = tid >> 1, ac = (tid & 1) * 16;
    const int br = tid;
    const __bf16* Ag = A + (size_t)(m0 + ar) * K + ac;
    const __bf16* Bg = B + (size_t)(n0 + br) * K;

    v8f acc[4][4];
    #pragma unroll
    for (int i = 0; i < 4; ++i)
        #pragma unroll
        for (int j = 0; j < 4; ++j) acc[i][j] = (v8f){0,0,0,0,0,0,0,0};

    // prologue: stage K-slice 0 into buffer 0
    {
        v16bf ra  = *(const v16bf*)(Ag);
        v16bf rb0 = *(const v16bf*)(Bg);
        v16bf rb1 = *(const v16bf*)(Bg + 16);
        *(v16bf*)&sA[0][ar][ac] = ra;
        *(v16bf*)&sB[0][br][0]  = rb0;
        *(v16bf*)&sB[0][br][16] = rb1;
    }
    __syncthreads();

    int buf = 0;
    for (int k0 = 0; k0 < K; k0 += 32) {
        const int nxt = k0 + 32;
        v16bf ra, rb0, rb1;
        if (nxt < K) {  // issue next slice's global loads early (overlap w/ WMMA)
            ra  = *(const v16bf*)(Ag + nxt);
            rb0 = *(const v16bf*)(Bg + nxt);
            rb1 = *(const v16bf*)(Bg + nxt + 16);
            __builtin_prefetch(Ag + nxt + 32, 0, 1);
        }

        // 16 WMMAs from current buffer
        v16bf af[4], bfr[4];
        #pragma unroll
        for (int i = 0; i < 4; ++i)
            af[i] = *(const v16bf*)&sA[buf][wm * 64 + i * 16 + row][kq];
        #pragma unroll
        for (int j = 0; j < 4; ++j)
            bfr[j] = *(const v16bf*)&sB[buf][wn * 64 + j * 16 + row][kq];
        #pragma unroll
        for (int i = 0; i < 4; ++i)
            #pragma unroll
            for (int j = 0; j < 4; ++j)
                acc[i][j] = __builtin_amdgcn_wmma_f32_16x16x32_bf16(
                    false, af[i], false, bfr[j], (short)0, acc[i][j], false, false);

        if (nxt < K) {  // stage next slice into the other buffer, single barrier
            *(v16bf*)&sA[buf ^ 1][ar][ac] = ra;
            *(v16bf*)&sB[buf ^ 1][br][0]  = rb0;
            *(v16bf*)&sB[buf ^ 1][br][16] = rb1;
            __syncthreads();
        }
        buf ^= 1;
    }

    // D layout: vgpr r, lanes 0-15 -> M=r, lanes 16-31 -> M=8+r; N = lane&15
    #pragma unroll
    for (int i = 0; i < 4; ++i) {
        #pragma unroll
        for (int j = 0; j < 4; ++j) {
            int mb = m0 + wm * 64 + i * 16 + (lane >> 4) * 8;
            int nn = n0 + wn * 64 + j * 16 + (lane & 15);
            #pragma unroll
            for (int r = 0; r < 8; ++r) {
                size_t idx = (size_t)(mb + r) * N + nn;
                float v = acc[i][j][r];
                if (Cb) Cb[idx] = (__bf16)v;
                else    Cf[idx] = v + (bias ? bias[nn] : 0.0f);
            }
        }
    }
}

// ---------------------------------------------------------------------------
// Local block attention: one workgroup per (batch, block, head).
// 4 waves; wave w handles query tiles t = w, w+4, w+8 (t<9, rows 16t..16t+15).
// scores: Q(16x64) @ K^T(64x160) -> softmax -> P(16x160) @ V(160x64).
// ---------------------------------------------------------------------------
__global__ __launch_bounds__(128)
void attn_kernel(const __bf16* __restrict__ qkv, __bf16* __restrict__ O) {
    __shared__ __align__(32) __bf16 sK [KPAD][DK];    // 20 KB  [key][feat]
    __shared__ __align__(32) __bf16 sVt[DK][KPAD];    // 20 KB  [feat][key]
    __shared__ __align__(32) __bf16 sP [4][16][KPAD]; // 20 KB  per-wave probs

    const int bid = blockIdx.x;               // b*NB*HEADS + blk*HEADS + h
    const int h   = bid % HEADS;
    const int blk = (bid / HEADS) % NB;
    const int b   = bid / (HEADS * NB);
    const size_t tokbase = (size_t)b * NTOK + (size_t)blk * BLKSZ;

    const int tid = threadIdx.x;
    // stage K (zero-padded to 160 rows)
    for (int e = tid; e < KPAD * (DK / 16); e += 128) {
        int r = e >> 2, c = (e & 3) * 16;
        v16bf val = {};
        if (r < BLKSZ)
            val = *(const v16bf*)(qkv + (tokbase + r) * QKV_LD + 1024 + h * DK + c);
        *(v16bf*)&sK[r][c] = val;
    }
    // stage V transposed [feat][key]
    for (int e = tid; e < DK * KPAD; e += 128) {
        int d = e / KPAD, r = e % KPAD;
        __bf16 val = (__bf16)0.0f;
        if (r < BLKSZ) val = qkv[(tokbase + r) * QKV_LD + 2048 + h * DK + d];
        sVt[d][r] = val;
    }
    __syncthreads();

    const int w    = tid >> 5;
    const int lane = tid & 31;
    const int row  = lane & 15;
    const int kq   = (lane >> 4) * 16;
    const float scale = 0.125f;               // 1/sqrt(DK)

    for (int t = w; t < 9; t += 4) {
        int mr = t * 16 + row; if (mr > BLKSZ - 1) mr = BLKSZ - 1;  // clamp pad rows
        const __bf16* qrow = qkv + (tokbase + mr) * QKV_LD + h * DK;

        v8f s[9];
        #pragma unroll
        for (int j = 0; j < 9; ++j) s[j] = (v8f){0,0,0,0,0,0,0,0};

        #pragma unroll
        for (int kt = 0; kt < DK; kt += 32) {
            v16bf a = *(const v16bf*)(qrow + kt + kq);
            #pragma unroll
            for (int j = 0; j < 9; ++j) {
                v16bf bb = *(const v16bf*)&sK[j * 16 + row][kt + kq];
                s[j] = __builtin_amdgcn_wmma_f32_16x16x32_bf16(
                    false, a, false, bb, (short)0, s[j], false, false);
            }
        }

        // scale + key mask + row softmax (rows r and r+8 split by lane group)
        float rmax[8], rsum[8];
        #pragma unroll
        for (int r = 0; r < 8; ++r) rmax[r] = -1e30f;
        #pragma unroll
        for (int j = 0; j < 9; ++j) {
            int col = j * 16 + (lane & 15);
            #pragma unroll
            for (int r = 0; r < 8; ++r) {
                float v = s[j][r] * scale;
                if (col >= BLKSZ) v = -1e30f;
                s[j][r] = v;
                rmax[r] = fmaxf(rmax[r], v);
            }
        }
        #pragma unroll
        for (int off = 1; off < 16; off <<= 1)
            #pragma unroll
            for (int r = 0; r < 8; ++r)
                rmax[r] = fmaxf(rmax[r], __shfl_xor(rmax[r], off, 32));
        #pragma unroll
        for (int r = 0; r < 8; ++r) rsum[r] = 0.0f;
        #pragma unroll
        for (int j = 0; j < 9; ++j)
            #pragma unroll
            for (int r = 0; r < 8; ++r) {
                float e = __expf(s[j][r] - rmax[r]);
                s[j][r] = e;
                rsum[r] += e;
            }
        #pragma unroll
        for (int off = 1; off < 16; off <<= 1)
            #pragma unroll
            for (int r = 0; r < 8; ++r)
                rsum[r] += __shfl_xor(rsum[r], off, 32);

        // stage probabilities to LDS (D-layout -> A-fragment layout reshape)
        #pragma unroll
        for (int j = 0; j < 9; ++j) {
            int col = j * 16 + (lane & 15);
            #pragma unroll
            for (int r = 0; r < 8; ++r)
                sP[w][r + 8 * (lane >> 4)][col] = (__bf16)s[j][r];
        }
        {   // zero pad cols 144..159
            int col = 144 + (lane & 15);
            #pragma unroll
            for (int r = 0; r < 8; ++r)
                sP[w][r + 8 * (lane >> 4)][col] = (__bf16)0.0f;
        }

        // P @ V
        v8f o[4];
        #pragma unroll
        for (int j = 0; j < 4; ++j) o[j] = (v8f){0,0,0,0,0,0,0,0};
        #pragma unroll
        for (int kt = 0; kt < KPAD; kt += 32) {
            v16bf a = *(const v16bf*)&sP[w][row][kt + kq];
            #pragma unroll
            for (int j = 0; j < 4; ++j) {
                v16bf bb = *(const v16bf*)&sVt[j * 16 + row][kt + kq];
                o[j] = __builtin_amdgcn_wmma_f32_16x16x32_bf16(
                    false, a, false, bb, (short)0, o[j], false, false);
            }
        }

        // normalize + store bf16
        #pragma unroll
        for (int j = 0; j < 4; ++j) {
            int n = j * 16 + (lane & 15);
            #pragma unroll
            for (int r = 0; r < 8; ++r) {
                int m = t * 16 + r + 8 * (lane >> 4);
                if (m < BLKSZ) {
                    float v = o[j][r] / (rsum[r] + 1e-20f);
                    O[(tokbase + m) * DIMX + h * DK + n] = (__bf16)v;
                }
            }
        }
    }
}

// ---------------------------------------------------------------------------
// Global attention over the 32 block-leader tokens; adds into row-0 outputs.
// One wave per (batch, head); lane = global query index.
// ---------------------------------------------------------------------------
__global__ __launch_bounds__(32)
void global_attn_kernel(const __bf16* __restrict__ qkv, __bf16* __restrict__ O) {
    const int b = blockIdx.x / HEADS, h = blockIdx.x % HEADS;
    const int g = threadIdx.x;                    // 0..31
    const size_t base = (size_t)b * NTOK;

    float q[DK];
    const __bf16* qr = qkv + (base + (size_t)g * BLKSZ) * QKV_LD + h * DK;
    #pragma unroll
    for (int d = 0; d < DK; ++d) q[d] = (float)qr[d];

    float sc[NB];
    float m = -1e30f;
    for (int kk = 0; kk < NB; ++kk) {
        const __bf16* kr = qkv + (base + (size_t)kk * BLKSZ) * QKV_LD + 1024 + h * DK;
        float dot = 0.0f;
        #pragma unroll
        for (int d = 0; d < DK; ++d) dot += q[d] * (float)kr[d];
        sc[kk] = dot * 0.125f;
        m = fmaxf(m, sc[kk]);
    }
    float sum = 0.0f;
    for (int kk = 0; kk < NB; ++kk) { sc[kk] = __expf(sc[kk] - m); sum += sc[kk]; }
    float inv = 1.0f / (sum + 1e-20f);

    __bf16* orow = O + (base + (size_t)g * BLKSZ) * DIMX + h * DK;
    for (int d = 0; d < DK; ++d) {
        float acc = 0.0f;
        for (int kk = 0; kk < NB; ++kk)
            acc += sc[kk] * (float)qkv[(base + (size_t)kk * BLKSZ) * QKV_LD + 2048 + h * DK + d];
        orow[d] = (__bf16)((float)orow[d] + acc * inv);
    }
}

// ---------------------------------------------------------------------------
// Host launcher
// ---------------------------------------------------------------------------
extern "C" void kernel_launch(void* const* d_in, const int* in_sizes, int n_in,
                              void* d_out, int out_size, void* d_ws, size_t ws_size,
                              hipStream_t stream) {
    const float* x  = (const float*)d_in[0];
    const float* Wq = (const float*)d_in[1];
    const float* Wk = (const float*)d_in[2];
    const float* Wv = (const float*)d_in[3];
    const float* Wo = (const float*)d_in[4];
    const float* bo = (const float*)d_in[5];

    // workspace layout (bf16 scratch), all 32B-aligned
    char* p = (char*)d_ws;
    __bf16* xb   = (__bf16*)p;  p += (size_t)MTOK * DIMX * 2;          // 33.8 MB
    __bf16* wcat = (__bf16*)p;  p += (size_t)3 * DIMX * DIMX * 2;      //  6.3 MB
    __bf16* wob  = (__bf16*)p;  p += (size_t)DIMX * DIMX * 2;          //  2.1 MB
    __bf16* qkv  = (__bf16*)p;  p += (size_t)MTOK * QKV_LD * 2;        // 101.4 MB
    __bf16* Obuf = (__bf16*)p;                                          // 33.8 MB

    // 1) downcast activations + weights to bf16
    cast_f32_bf16<<<2048, 256, 0, stream>>>(x,  xb,   (size_t)MTOK * DIMX);
    cast_f32_bf16<<<256,  256, 0, stream>>>(Wq, wcat,                       (size_t)DIMX * DIMX);
    cast_f32_bf16<<<256,  256, 0, stream>>>(Wk, wcat + (size_t)DIMX*DIMX,   (size_t)DIMX * DIMX);
    cast_f32_bf16<<<256,  256, 0, stream>>>(Wv, wcat + (size_t)2*DIMX*DIMX, (size_t)DIMX * DIMX);
    cast_f32_bf16<<<256,  256, 0, stream>>>(Wo, wob,                        (size_t)DIMX * DIMX);

    // 2) fused QKV projection: [16512,1024] @ [3072,1024]^T -> bf16
    gemm_bf16_wmma<<<dim3(MTOK / 128, QKV_LD / 256), 256, 0, stream>>>(
        xb, wcat, nullptr, qkv, nullptr, MTOK, QKV_LD, DIMX);

    // 3) block-local attention (2048 independent 129-token blocks)
    attn_kernel<<<BATCH * NB * HEADS, 128, 0, stream>>>(qkv, Obuf);

    // 4) global attention over block leaders, accumulate into row-0 outputs
    global_attn_kernel<<<BATCH * HEADS, 32, 0, stream>>>(qkv, Obuf);

    // 5) output projection: [16512,1024] @ [1024,1024]^T + bo -> f32 d_out
    gemm_bf16_wmma<<<dim3(MTOK / 128, DIMX / 256), 256, 0, stream>>>(
        Obuf, wob, (float*)d_out, nullptr, bo, MTOK, DIMX, DIMX);
}
